// MLA256Attention_65687229825560
// MI455X (gfx1250) — matrix operand order repaired and checked by reading, not compiled, not run on hardware
//
#include <hip/hip_runtime.h>

// ---------------------------------------------------------------------------
// Types / constants
// ---------------------------------------------------------------------------
typedef __attribute__((ext_vector_type(16))) __bf16 bf16x16;
typedef __attribute__((ext_vector_type(8)))  float  floatx8;
typedef __attribute__((ext_vector_type(4)))  unsigned int uint32x4;
typedef __attribute__((ext_vector_type(8)))  int  int32x8;
typedef __attribute__((ext_vector_type(4)))  int  int32x4;

#define B_SZ     4
#define T_SEQ    2048
#define D_MODEL  1024
#define N_HEADS  8
#define HEAD_DIM 256
#define KV_RANK  64
#define ROWS     (B_SZ * T_SEQ)          // 8192
#define QKV_DIM  (N_HEADS * HEAD_DIM)    // 2048

#if defined(__has_builtin)
#if __has_builtin(__builtin_amdgcn_tensor_load_to_lds) && \
    __has_builtin(__builtin_amdgcn_s_wait_tensorcnt)
#define HAVE_TDM 1
#else
#define HAVE_TDM 0
#endif
#else
#define HAVE_TDM 0
#endif

// ---------------------------------------------------------------------------
// Helpers
// ---------------------------------------------------------------------------
static __device__ __forceinline__ __bf16 f2bf(float f) {
    unsigned u = __builtin_bit_cast(unsigned, f);
    unsigned r = 0x7FFFu + ((u >> 16) & 1u);   // round-to-nearest-even
    u += r;
    unsigned short h = (unsigned short)(u >> 16);
    return __builtin_bit_cast(__bf16, h);
}

static __device__ __forceinline__ bf16x16 ldfrag2(const __bf16* p0, const __bf16* p1) {
    union { bf16x16 v; uint4 u[2]; } t;
    t.u[0] = *(const uint4*)p0;
    t.u[1] = *(const uint4*)p1;
    return t.v;
}
static __device__ __forceinline__ bf16x16 ldfrag16(const __bf16* p) {
    return ldfrag2(p, p + 8);
}

static __device__ __forceinline__ floatx8 wmma_bf16(bf16x16 a, bf16x16 b, floatx8 c) {
    return __builtin_amdgcn_wmma_f32_16x16x32_bf16(false, a, false, b, (short)0, c,
                                                   false, false);
}

static __device__ __forceinline__ float redmax16(float v) {
    #pragma unroll
    for (int m = 1; m < 16; m <<= 1) v = fmaxf(v, __shfl_xor(v, m, 32));
    return v;
}
static __device__ __forceinline__ float redsum16(float v) {
    #pragma unroll
    for (int m = 1; m < 16; m <<= 1) v += __shfl_xor(v, m, 32);
    return v;
}

#if HAVE_TDM
// TDM: load a 32(row) x 256(col) bf16 tile (row stride 2048 elements) from
// global into LDS, inserting 16B of padding after every 512B row so the LDS
// row stride becomes 264 elements (matches Ksh[32][264]).
static __device__ __forceinline__ void tdm_load_k_tile(unsigned lds_off,
                                                       const void* gptr) {
    unsigned long long ga = (unsigned long long)gptr;
    uint32x4 g0;
    g0[0] = 1u;                                   // count=1, user-mode load
    g0[1] = lds_off;                              // LDS byte address
    g0[2] = (unsigned)ga;                         // global_addr[31:0]
    g0[3] = (unsigned)((ga >> 32) & 0x1FFFFFFu)   // global_addr[56:32]
          | (2u << 30);                           // type = 2 ("image")
    int32x8 g1;
    g1[0] = (1 << 16)        // data_size = 2 bytes
          | (1 << 20)        // pad_enable
          | (6 << 22)        // pad_interval: every 128 DWORDs (512B)
          | (3 << 25);       // pad_amount: 4 DWORDs (16B)
    g1[1] = (int)(256u << 16);   // tensor_dim0[15:0]=256 (atomic_barrier_addr=0)
    g1[2] = (int)(32u  << 16);   // tensor_dim1[15:0]=32
    g1[3] = (int)(256u << 16);   // tile_dim0 = 256
    g1[4] = 32;                  // tile_dim1 = 32, tile_dim2 = 0
    g1[5] = 2048;                // tensor_dim0_stride[31:0] = 2048 elements
    g1[6] = 0;
    g1[7] = 0;
    int32x4 z4 = {};
#if __clang_major__ >= 23
    int32x8 z8 = {};
    __builtin_amdgcn_tensor_load_to_lds(g0, g1, z4, z4, z8, 0);
#else
    __builtin_amdgcn_tensor_load_to_lds(g0, g1, z4, z4, 0);
#endif
}
#endif

// ---------------------------------------------------------------------------
// f32 -> bf16 conversion
// ---------------------------------------------------------------------------
__global__ void cvt_f32_bf16(const float* __restrict__ in, __bf16* __restrict__ out, int n) {
    int i = blockIdx.x * blockDim.x + threadIdx.x;
    int stride = gridDim.x * blockDim.x;
    for (; i < n; i += stride) out[i] = f2bf(in[i]);
}

// ---------------------------------------------------------------------------
// GEMM: C[M,N] = A[M,K] * W[N,K]^T    (A, W bf16; C bf16 or f32)
// Block: 256 threads = 8 waves (2 x 4), block tile 64M x 256N, wave 32M x 64N.
// ---------------------------------------------------------------------------
__global__ __launch_bounds__(256)
void wmma_gemm_nt(const __bf16* __restrict__ A, const __bf16* __restrict__ W,
                  void* __restrict__ Cp, int M, int N, int K, int outBf16) {
    const int lane = threadIdx.x & 31;
    const int w    = threadIdx.x >> 5;
    const int l15  = lane & 15;
    const int lh   = lane >> 4;
    const int m0 = blockIdx.x * 64  + (w >> 2) * 32;
    const int n0 = blockIdx.y * 256 + (w & 3)  * 64;
    if (m0 >= M || n0 >= N) return;

    floatx8 acc[2][4] = {};

    const __bf16* arow[2];
    arow[0] = A + (size_t)(m0 + l15) * K;
    arow[1] = A + (size_t)(m0 + 16 + l15) * K;
    const __bf16* brow[4];
    #pragma unroll
    for (int j = 0; j < 4; ++j) brow[j] = W + (size_t)(n0 + 16 * j + l15) * K;

    const int koffA = lh * 8;   // A-frag: lanes 0-15 K=0..7/16..23, lanes 16-31 +8
    const int koffB = lh * 8;   // B-frag: lanes 0-15 K=0..7/16..23, lanes 16-31 +8

    for (int k0 = 0; k0 < K; k0 += 32) {
        if (k0 + 32 < K) {
            __builtin_prefetch(arow[0] + k0 + 32 + koffA, 0, 1);
            __builtin_prefetch(brow[0] + k0 + 32 + koffB, 0, 1);
        }
        bf16x16 a[2], b[4];
        #pragma unroll
        for (int i = 0; i < 2; ++i) {
            const __bf16* p = arow[i] + k0 + koffA;
            a[i] = ldfrag2(p, p + 16);
        }
        #pragma unroll
        for (int j = 0; j < 4; ++j)
            b[j] = ldfrag2(brow[j] + k0 + koffB, brow[j] + k0 + koffB + 16);
        #pragma unroll
        for (int i = 0; i < 2; ++i)
            #pragma unroll
            for (int j = 0; j < 4; ++j)
                acc[i][j] = wmma_bf16(a[i], b[j], acc[i][j]);
    }

    // C/D layout: VGPR r -> row r (lanes 0-15) / row r+8 (lanes 16-31), col = lane&15
    if (outBf16) {
        __bf16* C = (__bf16*)Cp;
        #pragma unroll
        for (int i = 0; i < 2; ++i)
            #pragma unroll
            for (int j = 0; j < 4; ++j)
                #pragma unroll
                for (int r = 0; r < 8; ++r) {
                    size_t row = (size_t)(m0 + 16 * i + lh * 8 + r);
                    C[row * N + n0 + 16 * j + l15] = f2bf(acc[i][j][r]);
                }
    } else {
        float* C = (float*)Cp;
        #pragma unroll
        for (int i = 0; i < 2; ++i)
            #pragma unroll
            for (int j = 0; j < 4; ++j)
                #pragma unroll
                for (int r = 0; r < 8; ++r) {
                    size_t row = (size_t)(m0 + 16 * i + lh * 8 + r);
                    C[row * N + n0 + 16 * j + l15] = acc[i][j][r];
                }
    }
}

// ---------------------------------------------------------------------------
// Flash attention, one (b,h) per blockIdx.x, 64 query rows per block
// (4 waves x 16 rows). Keys processed in tiles of 32 staged through LDS.
// K tile staged by the Tensor Data Mover (TENSORcnt); V tile transposed
// cooperatively on store.
// ---------------------------------------------------------------------------
__global__ __launch_bounds__(128)
void fa_mla(const __bf16* __restrict__ Q, const __bf16* __restrict__ Kg,
            const __bf16* __restrict__ Vg, __bf16* __restrict__ Og) {
    __shared__ __bf16 Ksh[32][264];     // key tile, row-major, padded stride
    __shared__ __bf16 VT[256][40];      // value tile transposed [feature][key]
    __shared__ __bf16 Psh[4][16][40];   // per-wave P round-trip buffer

    const int lane = threadIdx.x & 31;
    const int wave = threadIdx.x >> 5;
    const int l15  = lane & 15;
    const int lh   = lane >> 4;
    const int bh = blockIdx.x;
    const int b  = bh >> 3;
    const int h  = bh & 7;
    const int q0 = blockIdx.y * 64 + wave * 16;

    const size_t rs = QKV_DIM; // row stride of q/k/v/out (2048)
    const size_t qbase = ((size_t)b * T_SEQ + q0) * rs + (size_t)h * HEAD_DIM;

#if HAVE_TDM
    const unsigned ksh_lds = (unsigned)(unsigned long long)(void*)&Ksh[0][0];
#endif

    // Resident Q fragments: 16 x 256 as 8 A-frags (K chunks of 32)
    bf16x16 qf[8];
    #pragma unroll
    for (int ct = 0; ct < 8; ++ct) {
        const __bf16* p = Q + qbase + (size_t)l15 * rs + ct * 32 + lh * 8;
        qf[ct] = ldfrag2(p, p + 16);
    }

    floatx8 o[16] = {};
    float m_run[8], l_run[8];
    #pragma unroll
    for (int r = 0; r < 8; ++r) { m_run[r] = -3.0e38f; l_run[r] = 0.0f; }

    for (int kt = 0; kt < T_SEQ; kt += 32) {
        __syncthreads();
#if HAVE_TDM
        // One wave kicks off the TDM copy of the K tile (EXEC-independent,
        // per-wave op tracked with TENSORcnt).
        if (wave == 0) {
            tdm_load_k_tile(ksh_lds,
                            Kg + ((size_t)b * T_SEQ + kt) * rs + (size_t)h * HEAD_DIM);
        }
#endif
        // Cooperative load of the V tile, transposed into LDS.
        #pragma unroll
        for (int it = 0; it < 8; ++it) {
            int id = threadIdx.x + it * 128;   // 1024 chunks of 8 elements
            int r  = id >> 5;                  // key row 0..31
            int c  = (id & 31) * 8;            // feature 0..248
            size_t g = ((size_t)b * T_SEQ + kt + r) * rs + (size_t)h * HEAD_DIM + c;
#if !HAVE_TDM
            uint4 dk = *(const uint4*)(Kg + g);
            *(uint4*)&Ksh[r][c] = dk;
#endif
            uint4 dv = *(const uint4*)(Vg + g);
            const __bf16* e = (const __bf16*)&dv;
            #pragma unroll
            for (int x = 0; x < 8; ++x) VT[c + x][r] = e[x];
        }
#if HAVE_TDM
        __builtin_amdgcn_s_wait_tensorcnt(0);  // no-op for non-issuing waves
#endif
        __syncthreads();

        // S = Q @ K^T  (two 16x16 tiles covering 32 keys)
        floatx8 s0 = {}, s1 = {};
        #pragma unroll
        for (int ct = 0; ct < 8; ++ct) {
            bf16x16 bk0 = ldfrag2(&Ksh[l15][ct * 32 + lh * 8], &Ksh[l15][ct * 32 + lh * 8 + 16]);
            bf16x16 bk1 = ldfrag2(&Ksh[16 + l15][ct * 32 + lh * 8], &Ksh[16 + l15][ct * 32 + lh * 8 + 16]);
            s0 = wmma_bf16(qf[ct], bk0, s0);
            s1 = wmma_bf16(qf[ct], bk1, s1);
        }
        #pragma unroll
        for (int r = 0; r < 8; ++r) { s0[r] *= 0.0625f; s1[r] *= 0.0625f; }

        // Online softmax (per-row stats replicated across each 16-lane half)
        float corr[8];
        #pragma unroll
        for (int r = 0; r < 8; ++r) {
            float rm = redmax16(fmaxf(s0[r], s1[r]));
            float mn = fmaxf(m_run[r], rm);
            corr[r]  = __expf(m_run[r] - mn);
            m_run[r] = mn;
            float p0 = __expf(s0[r] - mn);
            float p1 = __expf(s1[r] - mn);
            Psh[wave][lh * 8 + r][l15]      = f2bf(p0);
            Psh[wave][lh * 8 + r][16 + l15] = f2bf(p1);
            l_run[r] = l_run[r] * corr[r] + redsum16(p0 + p1);
        }
        #pragma unroll
        for (int nt = 0; nt < 16; ++nt)
            #pragma unroll
            for (int r = 0; r < 8; ++r) o[nt][r] *= corr[r];

        // P back as A-fragment (C-layout -> A-layout via LDS; same-wave DS is in-order)
        const __bf16* pp = &Psh[wave][l15][lh * 8];
        bf16x16 pf = ldfrag2(pp, pp + 16);

        // O += P @ V   (V^T in LDS makes B-frags contiguous)
        #pragma unroll
        for (int nt = 0; nt < 16; ++nt) {
            bf16x16 bv = ldfrag2(&VT[nt * 16 + l15][lh * 8], &VT[nt * 16 + l15][lh * 8 + 16]);
            o[nt] = wmma_bf16(pf, bv, o[nt]);
        }
    }

    #pragma unroll
    for (int r = 0; r < 8; ++r) l_run[r] = 1.0f / l_run[r];
    #pragma unroll
    for (int nt = 0; nt < 16; ++nt)
        #pragma unroll
        for (int r = 0; r < 8; ++r) {
            size_t row = (size_t)b * T_SEQ + q0 + lh * 8 + r;
            Og[row * rs + (size_t)h * HEAD_DIM + nt * 16 + l15] =
                f2bf(o[nt][r] * l_run[r]);
        }
}

// ---------------------------------------------------------------------------
// Host launch
// ---------------------------------------------------------------------------
extern "C" void kernel_launch(void* const* d_in, const int* in_sizes, int n_in,
                              void* d_out, int out_size, void* d_ws, size_t ws_size,
                              hipStream_t stream) {
    (void)in_sizes; (void)n_in; (void)out_size; (void)ws_size;
    const float* x    = (const float*)d_in[0];
    const float* Wq   = (const float*)d_in[1];
    const float* Wkv  = (const float*)d_in[2];
    const float* Wku  = (const float*)d_in[3];
    const float* Wvu  = (const float*)d_in[4];
    const float* Wo   = (const float*)d_in[5];
    float* out = (float*)d_out;

    // Workspace layout (bytes)
    char* ws = (char*)d_ws;
    size_t off = 0;
    auto take = [&](size_t bytes) { char* p = ws + off; off += bytes; return p; };
    __bf16* xb    = (__bf16*)take((size_t)ROWS * D_MODEL * 2);
    __bf16* Wqb   = (__bf16*)take((size_t)QKV_DIM * D_MODEL * 2);
    __bf16* Wkvb  = (__bf16*)take((size_t)KV_RANK * D_MODEL * 2);
    __bf16* Wkub  = (__bf16*)take((size_t)QKV_DIM * KV_RANK * 2);
    __bf16* Wvub  = (__bf16*)take((size_t)QKV_DIM * KV_RANK * 2);
    __bf16* Wob   = (__bf16*)take((size_t)D_MODEL * QKV_DIM * 2);
    __bf16* qb    = (__bf16*)take((size_t)ROWS * QKV_DIM * 2);
    __bf16* latb  = (__bf16*)take((size_t)ROWS * KV_RANK * 2);
    __bf16* kb    = (__bf16*)take((size_t)ROWS * QKV_DIM * 2);
    __bf16* vb    = (__bf16*)take((size_t)ROWS * QKV_DIM * 2);
    __bf16* attnb = (__bf16*)take((size_t)ROWS * QKV_DIM * 2);

    auto cvt = [&](const float* in, __bf16* o2, int n) {
        int blocks = (n + 1023) / 1024;
        if (blocks > 8192) blocks = 8192;
        cvt_f32_bf16<<<blocks, 256, 0, stream>>>(in, o2, n);
    };
    cvt(x,   xb,   ROWS * D_MODEL);
    cvt(Wq,  Wqb,  QKV_DIM * D_MODEL);
    cvt(Wkv, Wkvb, KV_RANK * D_MODEL);
    cvt(Wku, Wkub, QKV_DIM * KV_RANK);
    cvt(Wvu, Wvub, QKV_DIM * KV_RANK);
    cvt(Wo,  Wob,  D_MODEL * QKV_DIM);

    auto gemm = [&](const __bf16* A, const __bf16* W, void* C,
                    int M, int N, int K, int outBf16) {
        dim3 grid(M / 64, (N + 255) / 256);
        wmma_gemm_nt<<<grid, 256, 0, stream>>>(A, W, C, M, N, K, outBf16);
    };

    // q = x @ Wq^T
    gemm(xb, Wqb, qb, ROWS, QKV_DIM, D_MODEL, 1);
    // kv_latent = x @ Wkv_down^T
    gemm(xb, Wkvb, latb, ROWS, KV_RANK, D_MODEL, 1);
    // k = latent @ Wk_up^T ; v = latent @ Wv_up^T
    gemm(latb, Wkub, kb, ROWS, QKV_DIM, KV_RANK, 1);
    gemm(latb, Wvub, vb, ROWS, QKV_DIM, KV_RANK, 1);

    // flash attention per (b,h)
    fa_mla<<<dim3(B_SZ * N_HEADS, T_SEQ / 64), 128, 0, stream>>>(qb, kb, vb, attnb);

    // out = attn @ Wo^T   (f32 output)
    gemm(attnb, Wob, out, ROWS, D_MODEL, QKV_DIM, 0);
}
